// MOEModel_28132035789064
// MI455X (gfx1250) — matrix operand order repaired and checked by reading, not compiled, not run on hardware
//
#include <hip/hip_runtime.h>

// ---------------------------------------------------------------------------
// CDNA5 (gfx1250, wave32) fused MoE pointer-network decoder.
// Matmuls: v_wmma_f32_16x16x32_bf16 (fp32 accum). Weight (B) tiles are
// pre-transposed to bf16 once, then streamed into LDS with
// global_load_async_to_lds_b128 (ASYNCcnt) -- the CDNA5 async copy path.
// ---------------------------------------------------------------------------

typedef __attribute__((ext_vector_type(16))) __bf16 v16bf;
typedef __attribute__((ext_vector_type(8)))  float  v8f;

union FragU {
    unsigned int u[8];
    v16bf b;
};

// pack two floats into two bf16 (round-half-up) with one v_perm_b32
__device__ __forceinline__ unsigned int pk_bf16(float lo, float hi) {
    union { float f; unsigned int u; } a, c;
    a.f = lo; c.f = hi;
    return __builtin_amdgcn_perm(c.u + 0x8000u, a.u + 0x8000u, 0x07060302u);
}
__device__ __forceinline__ unsigned short f2bf(float f) {
    union { float f; unsigned int u; } c; c.f = f;
    return (unsigned short)((c.u + 0x8000u) >> 16);
}

__device__ __forceinline__ v8f wmma_bf16(const FragU& a, const FragU& b, v8f c) {
    return __builtin_amdgcn_wmma_f32_16x16x32_bf16(
        false, a.b, false, b.b, (short)0, c, false, false);
}

// async 16-byte global -> LDS copy (per lane), tracked by ASYNCcnt
__device__ __forceinline__ void async_g2l_b128(unsigned lds_addr, const void* gptr) {
    unsigned long long ga = (unsigned long long)(size_t)gptr;
    asm volatile("global_load_async_to_lds_b128 %0, %1, off"
                 :: "v"(lds_addr), "v"(ga) : "memory");
}
__device__ __forceinline__ void wait_async0() {
    asm volatile("s_wait_asynccnt 0" ::: "memory");
}

#define NEG_BIG (-3.0e38f)

// ---------------------------------------------------------------------------
// One-time weight prep: WT[n][k] = bf16(W[k][n]) (n-major), zero-pad k >= K.
// grid.y = batch (experts), grid.x covers 256*ldk outputs.
// ---------------------------------------------------------------------------
__global__ __launch_bounds__(256)
void transpose_to_bf16(const float* __restrict__ W, unsigned short* __restrict__ WT,
                       int K, int ldk)
{
    const float* Wb = W + (size_t)blockIdx.y * K * 256;
    unsigned short* Tb = WT + (size_t)blockIdx.y * 256 * ldk;
    int o = blockIdx.x * 256 + threadIdx.x;
    if (o < 256 * ldk) {
        int n = o / ldk, k = o - n * ldk;
        float v = (k < K) ? Wb[k * 256 + n] : 0.f;
        Tb[o] = f2bf(v);
    }
}

// ---------------------------------------------------------------------------
// GEMM (K=N=256, M multiple of 32): C = A(Mx256,fp32) * B (pre-transposed
// bf16, n-major).  Tile 32x64, 8 waves, async B-tile copy to LDS.
// ---------------------------------------------------------------------------
__global__ __launch_bounds__(256)
void gemm_k256(const float* __restrict__ A, const unsigned short* __restrict__ BT,
               float* __restrict__ C)
{
    __shared__ unsigned short sA[32][36];
    __shared__ unsigned short sB[64][40];     // stride 40 -> 16B-aligned b128 rows
    const int tid = threadIdx.x, wave = tid >> 5, lane = tid & 31;
    const int tileM = blockIdx.y * 32, tileN = blockIdx.x * 64;
    const int wm = wave >> 2, wn = wave & 3;

    const int ar = tid >> 3, akq = (tid & 7) * 4;   // A fill: row, k-quad
    const int bnn = tid >> 2, bkc = (tid & 3) * 8;  // B fill: row, 8-elem k-chunk
    const float* aptr = A + (tileM + ar) * 256 + akq;
    const unsigned short* btp = BT + (tileN + bnn) * 256 + bkc;
    const unsigned lds_b = (unsigned)(size_t)&sB[bnn][bkc];

    const int arow = wm * 16 + (lane & 15), akb = (lane >> 4) * 8;
    const int bcol = wn * 16 + (lane & 15), bko = (lane >> 4) * 8;

    v8f acc = {};
    for (int kb = 0; kb < 256; kb += 32) {
        async_g2l_b128(lds_b, btp + kb);            // B tile: pure async DMA
        float4 av = *(const float4*)(aptr + kb);    // A tile: load+convert
        *(uint2*)&sA[ar][akq] = make_uint2(pk_bf16(av.x, av.y), pk_bf16(av.z, av.w));
        if (kb < 224) __builtin_prefetch(aptr + kb + 32, 0, 1);
        wait_async0();
        __syncthreads();
        FragU a, b;
#pragma unroll
        for (int vi = 0; vi < 4; ++vi)
            a.u[vi] = *(const unsigned int*)&sA[arow][akb + 2 * vi];
#pragma unroll
        for (int vi = 4; vi < 8; ++vi)
            a.u[vi] = *(const unsigned int*)&sA[arow][16 + akb + 2 * (vi - 4)];
#pragma unroll
        for (int vi = 0; vi < 8; ++vi)
            b.u[vi] = *(const unsigned int*)&sB[bcol][bko + 2 * vi + 2 * (vi & 4)];
        acc = wmma_bf16(a, b, acc);
        __syncthreads();
    }

    int n = tileN + wn * 16 + (lane & 15);
#pragma unroll
    for (int r = 0; r < 8; ++r) {
        int m = tileM + wm * 16 + r + 8 * (lane >> 4);
        C[m * 256 + n] = acc[r];
    }
}

// ---------------------------------------------------------------------------
// Q GEMM: C = [A|attr](25600x260) * Wq(260x256). WqT is zero-padded to
// K=288 (ldk=288) so the B path is uniform; A tail handles attr.
// ---------------------------------------------------------------------------
__global__ __launch_bounds__(256)
void gemm_q(const float* __restrict__ A, const float* __restrict__ attr,
            const unsigned short* __restrict__ BT, float* __restrict__ C)
{
    __shared__ unsigned short sA[32][36];
    __shared__ unsigned short sB[64][40];
    const int tid = threadIdx.x, wave = tid >> 5, lane = tid & 31;
    const int tileM = blockIdx.y * 32, tileN = blockIdx.x * 64;
    const int wm = wave >> 2, wn = wave & 3;

    const int ar = tid >> 3, akq = (tid & 7) * 4;
    const int bnn = tid >> 2, bkc = (tid & 3) * 8;
    const float* aptr = A + (tileM + ar) * 256 + akq;
    const unsigned short* btp = BT + (tileN + bnn) * 288 + bkc;
    const unsigned lds_b = (unsigned)(size_t)&sB[bnn][bkc];

    const int arow = wm * 16 + (lane & 15), akb = (lane >> 4) * 8;
    const int bcol = wn * 16 + (lane & 15), bko = (lane >> 4) * 8;

    v8f acc = {};
    for (int kb = 0; kb < 288; kb += 32) {
        async_g2l_b128(lds_b, btp + kb);
        uint2 za = make_uint2(0u, 0u);
        if (kb < 256) {                              // main: encoded_last_node
            float4 av = *(const float4*)(aptr + kb);
            za = make_uint2(pk_bf16(av.x, av.y), pk_bf16(av.z, av.w));
            if (kb < 224) __builtin_prefetch(aptr + kb + 32, 0, 1);
        } else if (akq == 0) {                       // tail: 4 attr columns
            float4 av = *(const float4*)(attr + (tileM + ar) * 4);
            za = make_uint2(pk_bf16(av.x, av.y), pk_bf16(av.z, av.w));
        }
        *(uint2*)&sA[ar][akq] = za;
        wait_async0();
        __syncthreads();
        FragU a, b;
#pragma unroll
        for (int vi = 0; vi < 4; ++vi)
            a.u[vi] = *(const unsigned int*)&sA[arow][akb + 2 * vi];
#pragma unroll
        for (int vi = 4; vi < 8; ++vi)
            a.u[vi] = *(const unsigned int*)&sA[arow][16 + akb + 2 * (vi - 4)];
#pragma unroll
        for (int vi = 0; vi < 8; ++vi)
            b.u[vi] = *(const unsigned int*)&sB[bcol][bko + 2 * vi + 2 * (vi & 4)];
        acc = wmma_bf16(a, b, acc);
        __syncthreads();
    }

    int n = tileN + wn * 16 + (lane & 15);
#pragma unroll
    for (int r = 0; r < 8; ++r) {
        int m = tileM + wm * 16 + r + 8 * (lane >> 4);
        C[m * 256 + n] = acc[r];
    }
}

// ---------------------------------------------------------------------------
// Fused MHA per (b,h): s = q k^T /4 + mask; softmax; out = p v.
// ---------------------------------------------------------------------------
__global__ __launch_bounds__(256)
void attn_kernel(const float* __restrict__ qg, const float* __restrict__ kg,
                 const float* __restrict__ vg, const float* __restrict__ ninf,
                 float* __restrict__ ctx)
{
    const int b = blockIdx.x >> 4;
    const int h = blockIdx.x & 15;
    const int tid = threadIdx.x, wave = tid >> 5, lane = tid & 31;

    __shared__ unsigned short sK[112][36];       // [n][d], d 16..31 zero
    __shared__ unsigned short sV[16][132];       // [d][n], n 101..127 zero
    __shared__ unsigned short sP[7][16][132];    // per-wave p transpose buffer

    {
        unsigned int* z = (unsigned int*)&sV[0][0];
        for (int i = tid; i < 16 * 132 / 2; i += 256) z[i] = 0u;
    }
    __syncthreads();
    for (int q = tid; q < 448; q += 256) {       // sK: n 0..111, d-quads
        int n = q >> 2, dq = (q & 3) * 4;
        uint2 val = make_uint2(0u, 0u);
        if (n < 101) {
            float4 kv = *(const float4*)(kg + (b * 101 + n) * 256 + h * 16 + dq);
            val = make_uint2(pk_bf16(kv.x, kv.y), pk_bf16(kv.z, kv.w));
        }
        *(uint2*)&sK[n][dq] = val;
        *(uint2*)&sK[n][16 + dq] = make_uint2(0u, 0u);
    }
    for (int q = tid; q < 404; q += 256) {       // sV: transpose v into [d][n]
        int n = q >> 2, dq = (q & 3) * 4;
        float4 vv = *(const float4*)(vg + (b * 101 + n) * 256 + h * 16 + dq);
        sV[dq + 0][n] = f2bf(vv.x);
        sV[dq + 1][n] = f2bf(vv.y);
        sV[dq + 2][n] = f2bf(vv.z);
        sV[dq + 3][n] = f2bf(vv.w);
    }
    __syncthreads();
    if (wave >= 7) return;

    const int pbase = wave * 16;

    FragU aq;
    {
        int p = pbase + (lane & 15);
        int dbase = (lane >> 4) * 8;
        aq.u[0] = aq.u[1] = aq.u[2] = aq.u[3] = 0;
        aq.u[4] = aq.u[5] = aq.u[6] = aq.u[7] = 0;   // K 16..31 zero pad
        if (p < 100) {
            const float* src = qg + ((b * 100 + p) * 256 + h * 16 + dbase);
            float4 q0 = *(const float4*)src;
            float4 q1 = *(const float4*)(src + 4);
            aq.u[0] = pk_bf16(q0.x, q0.y);
            aq.u[1] = pk_bf16(q0.z, q0.w);
            aq.u[2] = pk_bf16(q1.x, q1.y);
            aq.u[3] = pk_bf16(q1.z, q1.w);
        }
    }

    v8f s[7];
#pragma unroll
    for (int nt = 0; nt < 7; ++nt) {
        FragU bk;
        int col  = nt * 16 + (lane & 15);
        int koff = (lane >> 4) * 8;
#pragma unroll
        for (int vi = 0; vi < 8; ++vi)
            bk.u[vi] = *(const unsigned int*)&sK[col][koff + 2 * vi + 2 * (vi & 4)];
        v8f z = {};
        s[nt] = wmma_bf16(aq, bk, z);
    }

#pragma unroll
    for (int nt = 0; nt < 7; ++nt) {
        int n = nt * 16 + (lane & 15);
#pragma unroll
        for (int r = 0; r < 8; ++r) {
            int p = pbase + r + 8 * (lane >> 4);
            float val = s[nt][r] * 0.25f;
            bool pv = (p < 100), nv = (n < 101);
            if (pv && nv) val += ninf[(b * 100 + p) * 101 + n];
            s[nt][r] = pv ? (nv ? val : NEG_BIG) : 0.f;
        }
    }

#pragma unroll
    for (int r = 0; r < 8; ++r) {
        float mx = s[0][r];
#pragma unroll
        for (int nt = 1; nt < 7; ++nt) mx = fmaxf(mx, s[nt][r]);
#pragma unroll
        for (int off = 1; off < 16; off <<= 1)
            mx = fmaxf(mx, __shfl_xor(mx, off, 32));
        float sum = 0.f;
#pragma unroll
        for (int nt = 0; nt < 7; ++nt) {
            float e = __expf(s[nt][r] - mx);
            s[nt][r] = e;
            sum += e;
        }
#pragma unroll
        for (int off = 1; off < 16; off <<= 1)
            sum += __shfl_xor(sum, off, 32);
        float inv = 1.f / sum;
#pragma unroll
        for (int nt = 0; nt < 7; ++nt) s[nt][r] *= inv;
    }

#pragma unroll
    for (int nt = 0; nt < 7; ++nt) {
        int n = nt * 16 + (lane & 15);
#pragma unroll
        for (int r = 0; r < 8; ++r)
            sP[wave][r + 8 * (lane >> 4)][n] = f2bf(s[nt][r]);
    }
    {
        int n = 112 + (lane & 15);
#pragma unroll
        for (int r = 0; r < 8; ++r)
            sP[wave][r + 8 * (lane >> 4)][n] = 0;
    }
    asm volatile("s_wait_dscnt 0" ::: "memory");   // same-wave LDS RAW fence

    v8f o = {};
#pragma unroll
    for (int ks = 0; ks < 4; ++ks) {
        FragU ap, bv;
        int mrow  = lane & 15;
        int kbase = (lane >> 4) * 8;
#pragma unroll
        for (int vi = 0; vi < 4; ++vi)
            ap.u[vi] = *(const unsigned int*)&sP[wave][mrow][ks * 32 + kbase + 2 * vi];
#pragma unroll
        for (int vi = 4; vi < 8; ++vi)
            ap.u[vi] = *(const unsigned int*)&sP[wave][mrow][ks * 32 + 16 + kbase + 2 * (vi - 4)];
        int d    = lane & 15;
        int koff = (lane >> 4) * 8;
#pragma unroll
        for (int vi = 0; vi < 8; ++vi)
            bv.u[vi] = *(const unsigned int*)&sV[d][ks * 32 + koff + 2 * vi + 2 * (vi & 4)];
        o = wmma_bf16(ap, bv, o);
    }

    int d = lane & 15;
#pragma unroll
    for (int r = 0; r < 8; ++r) {
        int p = pbase + r + 8 * (lane >> 4);
        if (p < 100) ctx[(b * 100 + p) * 256 + h * 16 + d] = o[r];
    }
}

// ---------------------------------------------------------------------------
// Gate: logits = x*Wg (T x 8), top-2 softmax -> dense gates (T x 8).
// ---------------------------------------------------------------------------
__global__ __launch_bounds__(256)
void gate_kernel(const float* __restrict__ x, const float* __restrict__ Wg,
                 float* __restrict__ gates)
{
    const int wave = threadIdx.x >> 5, lane = threadIdx.x & 31;
    const int t = blockIdx.x * 8 + wave;
    float acc[8] = {};
#pragma unroll
    for (int it = 0; it < 2; ++it) {
        int k0 = it * 128 + lane * 4;
        float4 xv = *(const float4*)(x + t * 256 + k0);
        float xa[4] = {xv.x, xv.y, xv.z, xv.w};
#pragma unroll
        for (int j = 0; j < 4; ++j) {
            const float4* wr = (const float4*)(Wg + (k0 + j) * 8);
            float4 w0 = wr[0], w1 = wr[1];
            acc[0] += xa[j] * w0.x; acc[1] += xa[j] * w0.y;
            acc[2] += xa[j] * w0.z; acc[3] += xa[j] * w0.w;
            acc[4] += xa[j] * w1.x; acc[5] += xa[j] * w1.y;
            acc[6] += xa[j] * w1.z; acc[7] += xa[j] * w1.w;
        }
    }
#pragma unroll
    for (int e = 0; e < 8; ++e) {
#pragma unroll
        for (int off = 1; off < 32; off <<= 1)
            acc[e] += __shfl_xor(acc[e], off, 32);
    }
    if (lane == 0) {
        int i1 = 0;
#pragma unroll
        for (int e = 1; e < 8; ++e) if (acc[e] > acc[i1]) i1 = e;
        int i2 = (i1 == 0) ? 1 : 0;
#pragma unroll
        for (int e = 0; e < 8; ++e)
            if (e != i1 && acc[e] > acc[i2]) i2 = e;
        float g1 = 1.f / (1.f + __expf(acc[i2] - acc[i1]));
        float g2 = 1.f - g1;
#pragma unroll
        for (int e = 0; e < 8; ++e)
            gates[t * 8 + e] = (e == i1) ? g1 : ((e == i2) ? g2 : 0.f);
    }
}

// ---------------------------------------------------------------------------
// MoE combine: mh = sum_e diag(g_e) * x * W_e + sum_e g_e b_e.
// Experts pre-transposed to bf16 -> async B tiles; gate folded into A rows.
// ---------------------------------------------------------------------------
__global__ __launch_bounds__(256)
void moe_kernel(const float* __restrict__ x, const float* __restrict__ gates,
                const unsigned short* __restrict__ eWT, const float* __restrict__ eB,
                float* __restrict__ mh)
{
    __shared__ unsigned short sA[32][36];
    __shared__ unsigned short sB[64][40];
    __shared__ float sG[32][8];
    const int tid = threadIdx.x, wave = tid >> 5, lane = tid & 31;
    const int tileM = blockIdx.y * 32, tileN = blockIdx.x * 64;
    const int wm = wave >> 2, wn = wave & 3;

    const int ar = tid >> 3, akq = (tid & 7) * 4;
    const int bnn = tid >> 2, bkc = (tid & 3) * 8;
    const float* aptr = x + (tileM + ar) * 256 + akq;
    const unsigned lds_b = (unsigned)(size_t)&sB[bnn][bkc];

    const int arow = wm * 16 + (lane & 15), akb = (lane >> 4) * 8;
    const int bcol = wn * 16 + (lane & 15), bko = (lane >> 4) * 8;

    {
        int row = tid >> 3, e = tid & 7;
        sG[row][e] = gates[(tileM + row) * 8 + e];
    }
    __syncthreads();

    v8f acc = {};
    for (int e = 0; e < 8; ++e) {
        const unsigned short* btp = eWT + e * 65536 + (tileN + bnn) * 256 + bkc;
        const float g = sG[ar][e];
        for (int kb = 0; kb < 256; kb += 32) {
            async_g2l_b128(lds_b, btp + kb);
            float4 av = *(const float4*)(aptr + kb);
            *(uint2*)&sA[ar][akq] =
                make_uint2(pk_bf16(av.x * g, av.y * g), pk_bf16(av.z * g, av.w * g));
            wait_async0();
            __syncthreads();
            FragU a, b;
#pragma unroll
            for (int vi = 0; vi < 4; ++vi)
                a.u[vi] = *(const unsigned int*)&sA[arow][akb + 2 * vi];
#pragma unroll
            for (int vi = 4; vi < 8; ++vi)
                a.u[vi] = *(const unsigned int*)&sA[arow][16 + akb + 2 * (vi - 4)];
#pragma unroll
            for (int vi = 0; vi < 8; ++vi)
                b.u[vi] = *(const unsigned int*)&sB[bcol][bko + 2 * vi + 2 * (vi & 4)];
            acc = wmma_bf16(a, b, acc);
            __syncthreads();
        }
    }

    int n = tileN + wn * 16 + (lane & 15);
    float eb[8];
#pragma unroll
    for (int e = 0; e < 8; ++e) eb[e] = eB[e * 256 + n];
#pragma unroll
    for (int r = 0; r < 8; ++r) {
        int row = wm * 16 + r + 8 * (lane >> 4);
        float bias = 0.f;
#pragma unroll
        for (int e = 0; e < 8; ++e) bias += sG[row][e] * eb[e];
        mh[(tileM + row) * 256 + n] = acc[r] + bias;
    }
}

// ---------------------------------------------------------------------------
// Pointer: s2 = mh * enc^T + ab + ab1; 10*tanh(s2/16) + ninf; row softmax.
// ---------------------------------------------------------------------------
__global__ __launch_bounds__(256)
void pointer_kernel(const float* __restrict__ mh, const float* __restrict__ enc,
                    const float* __restrict__ ab, const float* __restrict__ ab1,
                    const float* __restrict__ ninf, float* __restrict__ probs)
{
    const int b = blockIdx.x;
    const int tid = threadIdx.x, wave = tid >> 5, lane = tid & 31;
    __shared__ unsigned short sB[112][36];   // enc tile, [n][k] (B^T: pairs along E)

    v8f acc[7] = {};
    const int p = wave * 16 + (lane & 15);
    const bool pv = (p < 100);
    const int kbase = (lane >> 4) * 8;

    for (int kb = 0; kb < 256; kb += 32) {
        for (int q = tid; q < 896; q += 256) {
            int n = q >> 3, kq = (q & 7) * 4;
            uint2 val = make_uint2(0u, 0u);
            if (n < 101) {
                float4 ev = *(const float4*)(enc + (b * 101 + n) * 256 + kb + kq);
                val = make_uint2(pk_bf16(ev.x, ev.y), pk_bf16(ev.z, ev.w));
            }
            *(uint2*)&sB[n][kq] = val;
        }
        __syncthreads();
        if (wave < 7) {
            FragU a;
            a.u[0] = a.u[1] = a.u[2] = a.u[3] = 0;
            a.u[4] = a.u[5] = a.u[6] = a.u[7] = 0;
            if (pv) {
                const float* ar2 = mh + (b * 100 + p) * 256 + kb;
                float4 a0 = *(const float4*)(ar2 + kbase);
                float4 a1 = *(const float4*)(ar2 + kbase + 4);
                float4 a2 = *(const float4*)(ar2 + 16 + kbase);
                float4 a3 = *(const float4*)(ar2 + 16 + kbase + 4);
                a.u[0] = pk_bf16(a0.x, a0.y); a.u[1] = pk_bf16(a0.z, a0.w);
                a.u[2] = pk_bf16(a1.x, a1.y); a.u[3] = pk_bf16(a1.z, a1.w);
                a.u[4] = pk_bf16(a2.x, a2.y); a.u[5] = pk_bf16(a2.z, a2.w);
                a.u[6] = pk_bf16(a3.x, a3.y); a.u[7] = pk_bf16(a3.z, a3.w);
            }
#pragma unroll
            for (int nt = 0; nt < 7; ++nt) {
                FragU bb;
                int col  = nt * 16 + (lane & 15);
                int koff = (lane >> 4) * 8;
#pragma unroll
                for (int vi = 0; vi < 8; ++vi)
                    bb.u[vi] = *(const unsigned int*)&sB[col][koff + 2 * vi + 2 * (vi & 4)];
                acc[nt] = wmma_bf16(a, bb, acc[nt]);
            }
        }
        __syncthreads();
    }
    if (wave >= 7) return;

    float s[7][8];
#pragma unroll
    for (int nt = 0; nt < 7; ++nt) {
        int n = nt * 16 + (lane & 15);
#pragma unroll
        for (int r = 0; r < 8; ++r) {
            int pr = wave * 16 + r + 8 * (lane >> 4);
            float val = NEG_BIG;
            if (pr < 100 && n < 101) {
                int idx = (b * 100 + pr) * 101 + n;
                float t = acc[nt][r] + ab[idx] + ab1[idx];
                val = 10.f * tanhf(t * 0.0625f) + ninf[idx];
            }
            if (pr >= 100) val = 0.f;
            s[nt][r] = val;
        }
    }
#pragma unroll
    for (int r = 0; r < 8; ++r) {
        float mx = s[0][r];
#pragma unroll
        for (int nt = 1; nt < 7; ++nt) mx = fmaxf(mx, s[nt][r]);
#pragma unroll
        for (int off = 1; off < 16; off <<= 1)
            mx = fmaxf(mx, __shfl_xor(mx, off, 32));
        float sum = 0.f;
#pragma unroll
        for (int nt = 0; nt < 7; ++nt) {
            float e = __expf(s[nt][r] - mx);
            s[nt][r] = e;
            sum += e;
        }
#pragma unroll
        for (int off = 1; off < 16; off <<= 1)
            sum += __shfl_xor(sum, off, 32);
        float inv = 1.f / sum;
#pragma unroll
        for (int nt = 0; nt < 7; ++nt) s[nt][r] *= inv;
    }
#pragma unroll
    for (int nt = 0; nt < 7; ++nt) {
        int n = nt * 16 + (lane & 15);
#pragma unroll
        for (int r = 0; r < 8; ++r) {
            int pr = wave * 16 + r + 8 * (lane >> 4);
            if (pr < 100 && n < 101)
                probs[(b * 100 + pr) * 101 + n] = s[nt][r];
        }
    }
}

// ---------------------------------------------------------------------------
extern "C" void kernel_launch(void* const* d_in, const int* in_sizes, int n_in,
                              void* d_out, int out_size, void* d_ws, size_t ws_size,
                              hipStream_t stream)
{
    (void)in_sizes; (void)n_in; (void)out_size; (void)ws_size;
    const float* last = (const float*)d_in[0];   // (256,100,256)
    const float* attr = (const float*)d_in[1];   // (256,100,4)
    const float* enc  = (const float*)d_in[2];   // (256,101,256)
    const float* ninf = (const float*)d_in[3];   // (256,100,101)
    const float* ab   = (const float*)d_in[4];   // (256,100,101)
    const float* ab1  = (const float*)d_in[5];   // (256,100,101)
    const float* Wq   = (const float*)d_in[6];   // (260,256)
    const float* Wk   = (const float*)d_in[7];   // (256,256)
    const float* Wv   = (const float*)d_in[8];   // (256,256)
    const float* Wg   = (const float*)d_in[9];   // (256,8)
    const float* eW   = (const float*)d_in[10];  // (8,256,256)
    const float* eB   = (const float*)d_in[11];  // (8,256)
    float* probs = (float*)d_out;                // (256,100,101)

    float* ws = (float*)d_ws;
    const size_t TQ = 25600ull * 256, TK = 25856ull * 256;
    float* q     = ws;                 // (25600,256)
    float* k     = q + TQ;             // (25856,256)
    float* v     = k + TK;             // (25856,256)
    float* ctx   = v + TK;             // (25600,256)
    float* gates = ctx + TQ;           // (25600,8)
    float* mh    = gates + 25600 * 8;  // (25600,256)
    // bf16 pre-transposed weights (n-major)
    unsigned short* WqT = (unsigned short*)(mh + TQ);  // 256 x 288 (K padded)
    unsigned short* WkT = WqT + 256 * 288;             // 256 x 256
    unsigned short* WvT = WkT + 65536;                 // 256 x 256
    unsigned short* eWT = WvT + 65536;                 // 8 x 256 x 256

    dim3 blk(256);
    transpose_to_bf16<<<dim3(288, 1), blk, 0, stream>>>(Wq, WqT, 260, 288);
    transpose_to_bf16<<<dim3(256, 1), blk, 0, stream>>>(Wk, WkT, 256, 256);
    transpose_to_bf16<<<dim3(256, 1), blk, 0, stream>>>(Wv, WvT, 256, 256);
    transpose_to_bf16<<<dim3(256, 8), blk, 0, stream>>>(eW, eWT, 256, 256);

    gemm_q   <<<dim3(4, 800), blk, 0, stream>>>(last, attr, WqT, q);
    gemm_k256<<<dim3(4, 808), blk, 0, stream>>>(enc, WkT, k);
    gemm_k256<<<dim3(4, 808), blk, 0, stream>>>(enc, WvT, v);
    attn_kernel<<<dim3(256 * 16), blk, 0, stream>>>(q, k, v, ninf, ctx);
    gate_kernel<<<dim3(25600 / 8), blk, 0, stream>>>(ctx, Wg, gates);
    moe_kernel<<<dim3(4, 800), blk, 0, stream>>>(ctx, gates, eWT, eB, mh);
    pointer_kernel<<<dim3(256), blk, 0, stream>>>(mh, enc, ab, ab1, ninf, probs);
}
